// DenseRelativeLoc_42142219108669
// MI455X (gfx1250) — matrix operand order repaired and checked by reading, not compiled, not run on hardware
//
#include <hip/hip_runtime.h>
#include <hip/hip_bf16.h>
#include <cstdint>
#include <cstddef>

// Problem constants (match reference)
constexpr int kB   = 256;
constexpr int kC   = 768;
constexpr int kHW  = 196;        // 14*14
constexpr int kS   = 196;
constexpr int kHID = 512;
constexpr int kM   = kB * kS;    // 50176 rows
constexpr int kK1  = 2 * kC;     // 1536

// Tiling: block 128x128, K-step 32; 8 waves as 4(M) x 2(N); wave tile 32x64
constexpr int BM = 128;
constexpr int BN = 128;
constexpr int BK = 32;
constexpr int LDA = BK + 8;      // 40 halfs = 80B row stride (16B aligned)
constexpr int LDB = BK + 8;

typedef __attribute__((ext_vector_type(16))) _Float16 v16h;
typedef __attribute__((ext_vector_type(8)))  _Float16 v8h;
typedef __attribute__((ext_vector_type(8)))  float    v8f;

// ---------------- weight cast/transpose kernels ----------------
// W1: [1536][512] f32  ->  W1t: [512][1536] f16   (B in [N][K] for WMMA frags)
__global__ void cast_w1_kernel(const float* __restrict__ W1, _Float16* __restrict__ W1t) {
  int id = blockIdx.x * blockDim.x + threadIdx.x;   // 1536*512 threads
  int k = id >> 9;          // /512
  int n = id & 511;
  W1t[(size_t)n * kK1 + k] = (_Float16)W1[id];
}
// W2: [512][512] f32 -> W2t: [512][512] f16 transposed
__global__ void cast_w2_kernel(const float* __restrict__ W2, _Float16* __restrict__ W2t) {
  int id = blockIdx.x * blockDim.x + threadIdx.x;   // 512*512 threads
  int k = id >> 9;
  int n = id & 511;
  W2t[(size_t)n * kHID + k] = (_Float16)W2[id];
}

// Shared compute step: 2 A-frags x 4 B-frags -> 8 WMMAs from LDS
__device__ __forceinline__
void wmma_step(const _Float16* sA, const _Float16* sB,
               int warow, int wbcol, int rl, int hi, v8f acc[2][4]) {
  const int c0 = hi * 8;        // A frag k-chunk 0
  const int c1 = 16 + hi * 8;   // A frag k-chunk 1
  const int cb = hi * 8;        // B frag k-chunk 0 (chunk 1 at +16)

  v16h a[2];
#pragma unroll
  for (int mi = 0; mi < 2; ++mi) {
    const int row = warow + mi * 16 + rl;
    v8h alo = *(const v8h*)&sA[row * LDA + c0];
    v8h ahi = *(const v8h*)&sA[row * LDA + c1];
#pragma unroll
    for (int i = 0; i < 8; ++i) { a[mi][i] = alo[i]; a[mi][i + 8] = ahi[i]; }
  }
  v16h bb[4];
#pragma unroll
  for (int j = 0; j < 4; ++j) {
    const int nrow = wbcol + j * 16 + rl;
    v8h blo = *(const v8h*)&sB[nrow * LDB + cb];
    v8h bhi = *(const v8h*)&sB[nrow * LDB + cb + 16];
#pragma unroll
    for (int i = 0; i < 8; ++i) { bb[j][i] = blo[i]; bb[j][i + 8] = bhi[i]; }
  }
#pragma unroll
  for (int mi = 0; mi < 2; ++mi)
#pragma unroll
    for (int j = 0; j < 4; ++j)
      acc[mi][j] = __builtin_amdgcn_wmma_f32_16x16x32_f16(
          false, a[mi], false, bb[j], (short)0, acc[mi][j], false, false);
}

// Shared epilogue: bias + relu -> f16 row-major [*, 512]
__device__ __forceinline__
void epilogue(v8f acc[2][4], const float* __restrict__ bias,
              _Float16* __restrict__ out, int rowBase, int colBase,
              int warow, int wbcol, int rl, int hi) {
#pragma unroll
  for (int mi = 0; mi < 2; ++mi) {
#pragma unroll
    for (int j = 0; j < 4; ++j) {
      const int n = colBase + wbcol + j * 16 + rl;
      const float bv = bias[n];
#pragma unroll
      for (int v = 0; v < 8; ++v) {
        const int m = rowBase + warow + mi * 16 + hi * 8 + v;
        float val = acc[mi][j][v] + bv;
        val = val > 0.f ? val : 0.f;
        out[(size_t)m * kHID + n] = (_Float16)val;
      }
    }
  }
}

// ---------------- GEMM1: fused gather + [M,1536]@[1536,512] + bias + relu ----------------
__global__ __launch_bounds__(256)
void gemm1_gather_kernel(const float* __restrict__ x,
                         const _Float16* __restrict__ Bt,   // W1t [512][1536]
                         const float* __restrict__ bias,    // b1 [512]
                         const int* __restrict__ pxs,       // [B,S,2]
                         const int* __restrict__ pys,
                         _Float16* __restrict__ out) {      // h1 [M][512]
  __shared__ _Float16 sA[BM * LDA];
  __shared__ _Float16 sB[BN * LDB];

  const int tid     = threadIdx.x;
  const int rowBase = blockIdx.x * BM;
  const int colBase = blockIdx.y * BN;

  // A-tile gather assignment: thread -> (row, 16 consecutive k)
  const int arow = tid & 127;
  const int akc  = (tid >> 7) << 4;          // 0,16
  const int gr   = rowBase + arow;           // global row in [0, M)
  const int bidx = gr / kS;
  const int ix   = pxs[gr * 2] * 14 + pxs[gr * 2 + 1];
  const int iy   = pys[gr * 2] * 14 + pys[gr * 2 + 1];
  const float* xb = x + (size_t)bidx * (kC * kHW);

  // B-tile load assignment: thread -> (n-row, 16 consecutive k)
  const int brow = tid & 127;
  const int bkc  = (tid >> 7) << 4;          // 0,16
  const _Float16* bg = Bt + (size_t)(colBase + brow) * kK1 + bkc;

  // per-wave WMMA assignment: 8 waves as 4 (M) x 2 (N), wave tile 32x64
  const int lane  = tid & 31;
  const int wave  = tid >> 5;
  const int rl    = lane & 15;
  const int hi    = lane >> 4;
  const int warow = (wave & 3) * 32;
  const int wbcol = (wave >> 2) * 64;

  v8f acc[2][4] = {};

  // register staging for double-buffered global traffic
  v8h aReg0, aReg1, bReg0, bReg1;

  auto fetch = [&](int k0) {
    // gather 16 f32 (channel-strided), cvt to f16
    const int kk  = k0 + akc;
    const int sec = (kk >= kC);
    const int ch  = sec ? (kk - kC) : kk;
    const int idx = sec ? iy : ix;
    const float* p = xb + (size_t)ch * kHW + idx;
#pragma unroll
    for (int j = 0; j < 8; ++j) aReg0[j] = (_Float16)p[(size_t)j * kHW];
#pragma unroll
    for (int j = 0; j < 8; ++j) aReg1[j] = (_Float16)p[(size_t)(j + 8) * kHW];
    const v8h* src = (const v8h*)(bg + k0);
    bReg0 = src[0];
    bReg1 = src[1];
  };

  fetch(0);
  for (int k0 = 0; k0 < kK1; k0 += BK) {
    // commit staged tile to LDS
    *(v8h*)&sA[arow * LDA + akc]     = aReg0;
    *(v8h*)&sA[arow * LDA + akc + 8] = aReg1;
    *(v8h*)&sB[brow * LDB + bkc]     = bReg0;
    *(v8h*)&sB[brow * LDB + bkc + 8] = bReg1;
    __syncthreads();
    // start next global fetch while computing from LDS
    if (k0 + BK < kK1) fetch(k0 + BK);
    wmma_step(sA, sB, warow, wbcol, rl, hi, acc);
    __syncthreads();
  }

  epilogue(acc, bias, out, rowBase, colBase, warow, wbcol, rl, hi);
}

// ---------------- GEMM2: [M,512]@[512,512] + bias + relu ----------------
__global__ __launch_bounds__(256)
void gemm2_kernel(const _Float16* __restrict__ A,   // h1 [M][512]
                  const _Float16* __restrict__ Bt,  // W2t [512][512]
                  const float* __restrict__ bias,   // b2
                  _Float16* __restrict__ out) {     // h2 [M][512]
  __shared__ _Float16 sA[BM * LDA];
  __shared__ _Float16 sB[BN * LDB];

  const int tid     = threadIdx.x;
  const int rowBase = blockIdx.x * BM;
  const int colBase = blockIdx.y * BN;

  const int arow = tid & 127;
  const int akc  = (tid >> 7) << 4;
  const _Float16* ag = A + (size_t)(rowBase + arow) * kHID + akc;

  const int brow = tid & 127;
  const int bkc  = (tid >> 7) << 4;
  const _Float16* bg = Bt + (size_t)(colBase + brow) * kHID + bkc;

  const int lane  = tid & 31;
  const int wave  = tid >> 5;
  const int rl    = lane & 15;
  const int hi    = lane >> 4;
  const int warow = (wave & 3) * 32;
  const int wbcol = (wave >> 2) * 64;

  v8f acc[2][4] = {};
  v8h aReg0, aReg1, bReg0, bReg1;

  auto fetch = [&](int k0) {
    const v8h* asrc = (const v8h*)(ag + k0);
    aReg0 = asrc[0];
    aReg1 = asrc[1];
    const v8h* bsrc = (const v8h*)(bg + k0);
    bReg0 = bsrc[0];
    bReg1 = bsrc[1];
  };

  fetch(0);
  for (int k0 = 0; k0 < kHID; k0 += BK) {
    *(v8h*)&sA[arow * LDA + akc]     = aReg0;
    *(v8h*)&sA[arow * LDA + akc + 8] = aReg1;
    *(v8h*)&sB[brow * LDB + bkc]     = bReg0;
    *(v8h*)&sB[brow * LDB + bkc + 8] = bReg1;
    __syncthreads();
    if (k0 + BK < kHID) fetch(k0 + BK);
    wmma_step(sA, sB, warow, wbcol, rl, hi, acc);
    __syncthreads();
  }

  epilogue(acc, bias, out, rowBase, colBase, warow, wbcol, rl, hi);
}

// ---------------- head: [M,512]@[512,2] + b3 ----------------
__global__ __launch_bounds__(256)
void head_kernel(const _Float16* __restrict__ h2,
                 const float* __restrict__ W3,   // [512][2]
                 const float* __restrict__ b3,   // [2]
                 float* __restrict__ pred) {     // [M][2]
  const int r = blockIdx.x * blockDim.x + threadIdx.x;   // kM threads
  const _Float16* hr = h2 + (size_t)r * kHID;
  float a0 = b3[0], a1 = b3[1];
  for (int k = 0; k < kHID; k += 8) {
    v8h hv = *(const v8h*)(hr + k);
#pragma unroll
    for (int i = 0; i < 8; ++i) {
      const float h = (float)hv[i];
      a0 += h * W3[(k + i) * 2 + 0];
      a1 += h * W3[(k + i) * 2 + 1];
    }
  }
  pred[r * 2 + 0] = a0;
  pred[r * 2 + 1] = a1;
}

// ---------------- deltaxy = (pxs - pys) + (H-1) ----------------
__global__ __launch_bounds__(256)
void delta_kernel(const int* __restrict__ pxs, const int* __restrict__ pys,
                  float* __restrict__ out2) {
  const int i = blockIdx.x * blockDim.x + threadIdx.x;   // kM*2 threads
  out2[i] = (float)(pxs[i] - pys[i]) + 13.0f;
}

extern "C" void kernel_launch(void* const* d_in, const int* in_sizes, int n_in,
                              void* d_out, int out_size, void* d_ws, size_t ws_size,
                              hipStream_t stream) {
  (void)in_sizes; (void)n_in; (void)out_size; (void)ws_size;

  const float* x  = (const float*)d_in[0];
  const float* W1 = (const float*)d_in[1];
  const float* b1 = (const float*)d_in[2];
  const float* W2 = (const float*)d_in[3];
  const float* b2 = (const float*)d_in[4];
  const float* W3 = (const float*)d_in[5];
  const float* b3 = (const float*)d_in[6];
  const int*   pxs = (const int*)d_in[7];
  const int*   pys = (const int*)d_in[8];

  // workspace layout (bytes):
  //   W1t f16: 512*1536*2 = 1,572,864   @ 0
  //   W2t f16: 512* 512*2 =   524,288   @ 1,572,864
  //   h1  f16: kM * 512*2 = 51,380,224  @ 2,097,152
  //   h2  f16: kM * 512*2 = 51,380,224  @ 53,477,376      (total 100 MB)
  char* ws = (char*)d_ws;
  _Float16* W1t = (_Float16*)(ws);
  _Float16* W2t = (_Float16*)(ws + 1572864);
  _Float16* h1  = (_Float16*)(ws + 2097152);
  _Float16* h2  = (_Float16*)(ws + 53477376);

  float* pred  = (float*)d_out;                  // [kM*2]
  float* delta = pred + (size_t)kM * 2;          // [kM*2]

  cast_w1_kernel<<<(kK1 * kHID) / 256, 256, 0, stream>>>(W1, W1t);
  cast_w2_kernel<<<(kHID * kHID) / 256, 256, 0, stream>>>(W2, W2t);

  dim3 g1(kM / BM, kHID / BN);                   // 392 x 4
  gemm1_gather_kernel<<<g1, 256, 0, stream>>>(x, W1t, b1, pxs, pys, h1);
  gemm2_kernel<<<g1, 256, 0, stream>>>(h1, W2t, b2, h2);

  head_kernel<<<kM / 256, 256, 0, stream>>>(h2, W3, b3, pred);
  delta_kernel<<<(kM * 2) / 256, 256, 0, stream>>>(pxs, pys, delta);
}
